// ccsm_3118146257688
// MI455X (gfx1250) — matrix operand order repaired and checked by reading, not compiled, not run on hardware
//
#include <hip/hip_runtime.h>

#define EPS 1e-5f
static constexpr int HW = 4096;   // 64*64
static constexpr int Bn = 8;      // batch

typedef __attribute__((ext_vector_type(16))) __bf16 v16bf;
typedef __attribute__((ext_vector_type(8)))  float  v8f;

struct B32x8 { uint4 lo, hi; };   // 32 bytes == one v16bf fragment

__device__ __forceinline__ unsigned short f2bf(float f) {
    unsigned int u = __builtin_bit_cast(unsigned int, f);
    u += 0x7FFFu + ((u >> 16) & 1u);          // round to nearest even
    return (unsigned short)(u >> 16);
}
__device__ __forceinline__ unsigned int pk2(float a, float b) {
    return (unsigned int)f2bf(a) | ((unsigned int)f2bf(b) << 16);
}
__device__ __forceinline__ float bf2f(unsigned short u) {
    return __builtin_bit_cast(float, (unsigned int)u << 16);
}
__device__ __forceinline__ float sigm(float x) { return 1.0f / (1.0f + __expf(-x)); }

// ---------------- GAP over H*W (f32 input): one block per (b,c) ----------------
__global__ void gap_kernel(const float* __restrict__ in, float* __restrict__ out) {
    int bc = blockIdx.x;
    const float* p = in + (size_t)bc * HW;
    float s = 0.f;
    for (int i = threadIdx.x; i < HW; i += 256) s += p[i];
    __shared__ float sh[256];
    sh[threadIdx.x] = s; __syncthreads();
    for (int off = 128; off > 0; off >>= 1) {
        if (threadIdx.x < off) sh[threadIdx.x] += sh[threadIdx.x + off];
        __syncthreads();
    }
    if (threadIdx.x == 0) out[bc] = sh[0] * (1.0f / HW);
}

// ---------------- GAP over H*W (bf16 input, packed pairs): one block per (b,c) ----------------
__global__ void gap_bf_kernel(const unsigned int* __restrict__ in, float* __restrict__ out) {
    int bc = blockIdx.x;
    const unsigned int* p = in + (size_t)bc * (HW / 2);
    float s = 0.f;
    for (int i = threadIdx.x; i < HW / 2; i += 256) {
        unsigned int u = p[i];
        s += bf2f((unsigned short)(u & 0xFFFFu)) + bf2f((unsigned short)(u >> 16));
    }
    __shared__ float sh[256];
    sh[threadIdx.x] = s; __syncthreads();
    for (int off = 128; off > 0; off >>= 1) {
        if (threadIdx.x < off) sh[threadIdx.x] += sh[threadIdx.x + off];
        __syncthreads();
    }
    if (threadIdx.x == 0) out[bc] = sh[0] * (1.0f / HW);
}

// ---------------- initial ChannelAttention: sigmoid(relu(gap @ ca_w^T)) ----------------
__global__ void ca_kernel(const float* __restrict__ gap, const float* __restrict__ ca_w,
                          float* __restrict__ chatt, int C) {
    int idx = blockIdx.x * 256 + threadIdx.x;      // over B*C
    int b = idx / C, c = idx % C;
    const float* g = gap + (size_t)b * C;
    const float* w = ca_w + (size_t)c * C;
    float s = 0.f;
    for (int i = 0; i < C; i++) s += g[i] * w[i];
    chatt[idx] = sigm(fmaxf(s, 0.f));
}

// ---------------- x1 = bf16(x * chatt), float4 -> packed bf16x4 ----------------
__global__ void cscale_bf_kernel(const float4* __restrict__ x, const float* __restrict__ chatt,
                                 uint2* __restrict__ out) {
    size_t idx = (size_t)blockIdx.x * 256 + threadIdx.x;   // over B*C*HW/4
    size_t chan = idx / (HW / 4);
    float s = chatt[chan];
    float4 v = x[idx];
    uint2 o;
    o.x = pk2(v.x * s, v.y * s);
    o.y = pk2(v.z * s, v.w * s);
    out[idx] = o;
}

// ---------------- ODConv attention: one block per sample; emits bf16 W' and fil ----------------
__global__ void odattn_kernel(const float* __restrict__ gap,
                              const float* __restrict__ fc_w,
                              const float* __restrict__ bng, const float* __restrict__ bnb,
                              const float* __restrict__ bnm, const float* __restrict__ bnv,
                              const float* __restrict__ ch_w, const float* __restrict__ ch_b,
                              const float* __restrict__ fil_w, const float* __restrict__ fil_b,
                              const float* __restrict__ ker_w, const float* __restrict__ ker_b,
                              const float* __restrict__ wbank,
                              unsigned short* __restrict__ Wb, float* __restrict__ fil_out,
                              int Cin, int Cout, int A) {
    int b = blockIdx.x, t = threadIdx.x;
    __shared__ float a_sh[32];
    __shared__ float ch_sh[512];
    __shared__ float kl[4];
    __shared__ float ker_sh[4];
    const float* g = gap + (size_t)b * Cin;

    if (t < A) {
        float s = 0.f;
        const float* w = fc_w + (size_t)t * Cin;
        for (int i = 0; i < Cin; i++) s += g[i] * w[i];
        s = (s - bnm[t]) * (bng[t] * rsqrtf(bnv[t] + EPS)) + bnb[t];
        a_sh[t] = fmaxf(s, 0.f);
    }
    __syncthreads();

    for (int i = t; i < Cin; i += 256) {
        float s = ch_b[i];
        const float* w = ch_w + (size_t)i * A;
        for (int j = 0; j < A; j++) s += a_sh[j] * w[j];
        ch_sh[i] = sigm(s);
    }
    for (int o = t; o < Cout; o += 256) {
        float s = fil_b[o];
        const float* w = fil_w + (size_t)o * A;
        for (int j = 0; j < A; j++) s += a_sh[j] * w[j];
        fil_out[(size_t)b * Cout + o] = sigm(s);
    }
    if (t < 4) {
        float s = ker_b[t];
        const float* w = ker_w + (size_t)t * A;
        for (int j = 0; j < A; j++) s += a_sh[j] * w[j];
        kl[t] = s;
    }
    __syncthreads();
    if (t < 4) {
        float mx = fmaxf(fmaxf(kl[0], kl[1]), fmaxf(kl[2], kl[3]));
        float den = __expf(kl[0] - mx) + __expf(kl[1] - mx) + __expf(kl[2] - mx) + __expf(kl[3] - mx);
        ker_sh[t] = __expf(kl[t] - mx) / den;
    }
    __syncthreads();

    int n = Cout * Cin;
    for (int idx = t; idx < n; idx += 256) {
        int i = idx % Cin;
        float v = ker_sh[0] * wbank[idx] + ker_sh[1] * wbank[n + idx]
                + ker_sh[2] * wbank[2 * n + idx] + ker_sh[3] * wbank[3 * n + idx];
        Wb[(size_t)b * n + idx] = f2bf(v * ch_sh[i]);   // fold channel attention into W'
    }
}

// ---------------- fold eval-mode BN into scale/shift ----------------
__global__ void bnp_kernel(const float* g, const float* bb, const float* m, const float* v,
                           float* esc, float* esh, int C) {
    int i = blockIdx.x * 256 + threadIdx.x;
    if (i < C) {
        float s = g[i] * rsqrtf(v[i] + EPS);
        esc[i] = s;
        esh[i] = bb[i] - s * m[i];
    }
}

// ---------------- batched GEMM: one wave computes a 64x64 tile via 4x4 WMMAs ----------------
// A (bf16 W'): b128 loads; B (bf16 activations): strided 16-bit loads; bit_cast, no conversion ALU in loop.
template <bool OUT_F32>
__global__ __launch_bounds__(32)
void od_gemm_kernel(const unsigned short* __restrict__ Wb,   // (B,Cout,Cin) bf16
                    const unsigned short* __restrict__ X,    // (B,Cin,HW) bf16
                    const float* __restrict__ fil,           // (B,Cout)
                    const float* __restrict__ esc, const float* __restrict__ esh,
                    const unsigned short* __restrict__ res,  // bf16 residual or null
                    void* __restrict__ OutP,                 // (B,Cout,HW) bf16 or f32
                    int Cin, int Cout) {
    const int lane = threadIdx.x;
    const int half = lane >> 4;
    const int l16  = lane & 15;
    const int n0 = blockIdx.x * 64;
    const int m0 = blockIdx.y * 64;
    const int b  = blockIdx.z;

    const unsigned short* Wbase = Wb + (size_t)b * Cout * Cin;
    const unsigned short* Xb = X + (size_t)b * Cin * HW;

    v8f acc[4][4];
    #pragma unroll
    for (int i = 0; i < 4; i++)
        #pragma unroll
        for (int j = 0; j < 4; j++)
            #pragma unroll
            for (int e = 0; e < 8; e++) acc[i][j][e] = 0.f;

    for (int k0 = 0; k0 < Cin; k0 += 32) {
        // A fragments (16x32 bf16): lane l16 = row; K = {8*half+0..7, 16+8*half+0..7}
        v16bf afrag[4];
        #pragma unroll
        for (int tm = 0; tm < 4; tm++) {
            const unsigned short* wr =
                Wbase + (size_t)(m0 + tm * 16 + l16) * Cin + k0 + 8 * half;
            B32x8 u;
            u.lo = *(const uint4*)(wr);
            u.hi = *(const uint4*)(wr + 16);
            afrag[tm] = __builtin_bit_cast(v16bf, u);
        }
        // B fragments (32x16 bf16): lane l16 = N column; K = {8*half+0..7, 16+8*half+0..7}
        v16bf bfrag[4];
        #pragma unroll
        for (int tn = 0; tn < 4; tn++) {
            const unsigned short* xr = Xb + (size_t)(k0 + 8 * half) * HW + n0 + tn * 16 + l16;
            unsigned short u[16];
            #pragma unroll
            for (int s = 0; s < 8; s++) {
                u[s]     = xr[(size_t)s * HW];
                u[8 + s] = xr[(size_t)(16 + s) * HW];
            }
            bfrag[tn] = __builtin_bit_cast(v16bf, u);
        }
        // prefetch next K-slab of B (rows stride 8KB; one line per lane)
        if (k0 + 32 < Cin)
            __builtin_prefetch(Xb + (size_t)(k0 + 32 + lane) * HW + n0, 0, 3);

        #pragma unroll
        for (int tm = 0; tm < 4; tm++)
            #pragma unroll
            for (int tn = 0; tn < 4; tn++)
                acc[tm][tn] = __builtin_amdgcn_wmma_f32_16x16x32_bf16(
                    false, afrag[tm], false, bfrag[tn],
                    (short)0, acc[tm][tn], false, false);
    }

    // epilogue: fil * acc -> relu -> BN(scale,shift) -> (+residual) -> store
    float* outf = (float*)OutP;
    unsigned short* outb = (unsigned short*)OutP;
    #pragma unroll
    for (int tm = 0; tm < 4; tm++) {
        #pragma unroll
        for (int r = 0; r < 8; r++) {
            int grow = m0 + tm * 16 + r + 8 * half;
            float fs = fil[(size_t)b * Cout + grow];
            float sc = esc[grow], sh = esh[grow];
            #pragma unroll
            for (int tn = 0; tn < 4; tn++) {
                int gcol = n0 + tn * 16 + l16;
                float v = fmaxf(acc[tm][tn][r] * fs, 0.f) * sc + sh;
                size_t oidx = ((size_t)b * Cout + grow) * HW + gcol;
                if (res) v += bf2f(res[oidx]);
                if (OUT_F32) outf[oidx] = v;
                else         outb[oidx] = f2bf(v);
            }
        }
    }
}

// ---------------- mean over channels (f32 input) ----------------
__global__ void chanmean_kernel(const float* __restrict__ y, float* __restrict__ m, int C) {
    int idx = blockIdx.x * 256 + threadIdx.x;    // over B*HW
    int b = idx / HW, p = idx % HW;
    const float* yp = y + (size_t)b * C * HW + p;
    float s = 0.f;
    for (int c = 0; c < C; c++) s += yp[(size_t)c * HW];
    m[idx] = s * (1.0f / C);
}

// ---------------- 7x7 spatial conv (pad 3) + sigmoid ----------------
__global__ void sa_kernel(const float* __restrict__ m, const float* __restrict__ sw,
                          float* __restrict__ s) {
    int idx = blockIdx.x * 256 + threadIdx.x;    // over B*HW
    int b = idx / HW, p = idx % HW;
    int h = p >> 6, w = p & 63;
    const float* mb = m + (size_t)b * HW;
    float a = 0.f;
    for (int kh = 0; kh < 7; kh++) {
        int hh = h + kh - 3;
        if (hh < 0 || hh >= 64) continue;
        for (int kw = 0; kw < 7; kw++) {
            int ww = w + kw - 3;
            if (ww < 0 || ww >= 64) continue;
            a += mb[hh * 64 + ww] * sw[kh * 7 + kw];
        }
    }
    s[idx] = sigm(a);
}

// ---------------- out *= smap (broadcast over channels), in place ----------------
__global__ void samul_kernel(float* __restrict__ out, const float* __restrict__ s, int C) {
    size_t idx = (size_t)blockIdx.x * 256 + threadIdx.x;   // over B*C*HW
    size_t b = idx / ((size_t)C * HW);
    size_t p = idx % HW;
    out[idx] *= s[b * HW + p];
}

extern "C" void kernel_launch(void* const* d_in, const int* in_sizes, int n_in,
                              void* d_out, int out_size, void* d_ws, size_t ws_size,
                              hipStream_t stream) {
    (void)in_sizes; (void)n_in; (void)out_size; (void)ws_size;
    int i = 0;
    auto nf = [&]() -> const float* { return (const float*)d_in[i++]; };

    const float* x    = nf();
    const float* ca_w = nf();
    const float* sa_w = nf();

    struct OD { const float *fc,*bg,*bb,*bm,*bv,*chw,*chb,*fw,*fb,*kw,*kb,*w; };
    struct BN { const float *g,*b,*m,*v; };
    OD od1{nf(),nf(),nf(),nf(),nf(),nf(),nf(),nf(),nf(),nf(),nf(),nf()};
    BN bn1{nf(),nf(),nf(),nf()};
    OD od2{nf(),nf(),nf(),nf(),nf(),nf(),nf(),nf(),nf(),nf(),nf(),nf()};
    BN bn2{nf(),nf(),nf(),nf()};
    OD od3{nf(),nf(),nf(),nf(),nf(),nf(),nf(),nf(),nf(),nf(),nf(),nf()};
    BN bn3{nf(),nf(),nf(),nf()};
    OD od4{nf(),nf(),nf(),nf(),nf(),nf(),nf(),nf(),nf(),nf(),nf(),nf()};
    BN bn4{nf(),nf(),nf(),nf()};

    // workspace layout
    char* ws = (char*)d_ws;
    size_t off = 0;
    auto alloc = [&](size_t bytes) -> void* {
        void* p = ws + off;
        off += (bytes + 255) & ~(size_t)255;
        return p;
    };
    unsigned short* bufA  = (unsigned short*)alloc((size_t)Bn * 512 * HW * 2); // 32 MB bf16
    unsigned short* bufB  = (unsigned short*)alloc((size_t)Bn * 512 * HW * 2); // 32 MB bf16
    unsigned short* Wb    = (unsigned short*)alloc((size_t)Bn * 512 * 512 * 2);
    float*          gapb  = (float*)alloc((size_t)Bn * 512 * 4);
    float*          chatt = (float*)alloc((size_t)Bn * 512 * 4);
    float*          filb  = (float*)alloc((size_t)Bn * 512 * 4);
    float*          esc   = (float*)alloc(512 * 4);
    float*          esh   = (float*)alloc(512 * 4);
    float*          mbuf  = (float*)alloc((size_t)Bn * HW * 4);
    float*          smap  = (float*)alloc((size_t)Bn * HW * 4);
    float*          yout  = (float*)d_out;

    // ---- initial channel attention (from f32 x) ----
    gap_kernel<<<Bn * 512, 256, 0, stream>>>(x, gapb);
    ca_kernel<<<(Bn * 512) / 256, 256, 0, stream>>>(gapb, ca_w, chatt, 512);
    cscale_bf_kernel<<<(Bn * 512 * HW / 4) / 256, 256, 0, stream>>>(
        (const float4*)x, chatt, (uint2*)bufA);

    auto run_block = [&](const unsigned short* in, int Cin, int Cout, int A,
                         const OD& od, const BN& bn, const unsigned short* resid,
                         void* out, bool out_f32) {
        gap_bf_kernel<<<Bn * Cin, 256, 0, stream>>>((const unsigned int*)in, gapb);
        odattn_kernel<<<Bn, 256, 0, stream>>>(gapb, od.fc, od.bg, od.bb, od.bm, od.bv,
                                              od.chw, od.chb, od.fw, od.fb, od.kw, od.kb,
                                              od.w, Wb, filb, Cin, Cout, A);
        bnp_kernel<<<(Cout + 255) / 256, 256, 0, stream>>>(bn.g, bn.b, bn.m, bn.v, esc, esh, Cout);
        dim3 grid(HW / 64, Cout / 64, Bn);
        if (out_f32)
            od_gemm_kernel<true><<<grid, 32, 0, stream>>>(Wb, in, filb, esc, esh,
                                                          resid, out, Cin, Cout);
        else
            od_gemm_kernel<false><<<grid, 32, 0, stream>>>(Wb, in, filb, esc, esh,
                                                           resid, out, Cin, Cout);
    };

    run_block(bufA, 512, 512, 32, od1, bn1, bufA,    bufB, false); // y1 = block1(x1)+x1
    run_block(bufB, 512, 256, 32, od2, bn2, nullptr, bufA, false); // y2 = block2(y1)
    run_block(bufA, 256, 256, 16, od3, bn3, bufA,    bufB, false); // y3 = block3(y2)+y2
    run_block(bufB, 256, 128, 16, od4, bn4, nullptr, yout, true);  // y4 -> d_out (f32)

    // ---- spatial attention ----
    chanmean_kernel<<<(Bn * HW) / 256, 256, 0, stream>>>(yout, mbuf, 128);
    sa_kernel<<<(Bn * HW) / 256, 256, 0, stream>>>(mbuf, sa_w, smap);
    samul_kernel<<<((size_t)Bn * 128 * HW) / 256, 256, 0, stream>>>(yout, smap, 128);
}
